// LSTMModel_32160715112738
// MI455X (gfx1250) — compile-verified, matrix operand order checked by reading, not run on hardware
//
#include <hip/hip_runtime.h>
#include <hip/hip_bf16.h>
#include <math.h>

typedef __attribute__((ext_vector_type(16))) _Float16 v16h;
typedef __attribute__((ext_vector_type(8)))  float    v8f;
typedef __attribute__((ext_vector_type(4)))  unsigned int v4u;
typedef __attribute__((ext_vector_type(8)))  int      v8i_;
typedef __attribute__((ext_vector_type(4)))  int      v4i_;

union Frag16 { v16h v; uint4 q[2]; };
union AccU   { v8f  v; float f[8]; };

#define BSZ 64
#define TLEN 2048
#define DIN 128
#define HID 256
#define OUTD 128
#define G4H 1024
#define XSTAGE_ELEMS (16 * G4H)          /* 16 batch rows x 4H f32 = 64 KB */

#if defined(__has_builtin)
#if __has_builtin(__builtin_amdgcn_tensor_load_to_lds) && __has_builtin(__builtin_amdgcn_s_wait_tensorcnt)
#define HAVE_TDM 1
#endif
#endif
#ifndef HAVE_TDM
#define HAVE_TDM 0
#endif

// ---------------------------------------------------------------------------
// f32 -> f16 elementwise convert (for X)
// ---------------------------------------------------------------------------
__global__ __launch_bounds__(256)
void cvt_f16_kernel(const float* __restrict__ in, _Float16* __restrict__ out, int n) {
    int i = blockIdx.x * blockDim.x + threadIdx.x;
    if (i < n) out[i] = (_Float16)in[i];
}

// ---------------------------------------------------------------------------
// Pack a row-major f32 weight [K x N] into WMMA B-fragment order, f16.
// Element e of lane L maps to K = ktile*32 + (L>>4)*8 + (e<8 ? e : e+8),
//                           N = ntile*16 + (L&15).
// ---------------------------------------------------------------------------
__global__ __launch_bounds__(256)
void pack_b_kernel(const float* __restrict__ W, _Float16* __restrict__ out,
                   int K, int N) {
    int nkt = K >> 5;
    int total = nkt * (N >> 4) * 32;
    int tid = blockIdx.x * blockDim.x + threadIdx.x;
    if (tid >= total) return;
    int lane = tid & 31;
    int tileIdx = tid >> 5;
    int ktile = tileIdx % nkt;
    int ntile = tileIdx / nkt;
    int kbase = ktile * 32 + (lane >> 4) * 8;
    int ncol  = ntile * 16 + (lane & 15);
    _Float16* o = out + (size_t)tid * 16;
#pragma unroll
    for (int e = 0; e < 16; ++e) {
        int k = kbase + (e < 8 ? e : e + 8);
        o[e] = (_Float16)W[(size_t)k * N + ncol];
    }
}

// ---------------------------------------------------------------------------
// GEMM: A[M x K] (f16 row-major, M enumerated as b*T+t) * Bp (packed f16)
//       + bias[N] -> f32.
// mode 0: C row-major [M x N] with row = m = b*T + t
// mode 1: C is [T x B x N] : row' = t*B + b  (for TDM-friendly xg layout)
// ---------------------------------------------------------------------------
__global__ __launch_bounds__(256)
void gemm_wmma_kernel(const _Float16* __restrict__ A,
                      const _Float16* __restrict__ Bp,
                      const float* __restrict__ bias,
                      float* __restrict__ C, int K, int N, int mode) {
    const int lane  = threadIdx.x & 31;
    const int wave  = threadIdx.x >> 5;
    const int mtile = blockIdx.x;
    const int ntile = blockIdx.y * 8 + wave;
    const int nkt   = K >> 5;
    const int hi    = lane >> 4;
    const int lo    = lane & 15;
    const int col   = ntile * 16 + lo;

    AccU acc;
    float bv = bias[col];
#pragma unroll
    for (int j = 0; j < 8; ++j) acc.f[j] = bv;

    const _Float16* arp = A + (size_t)(mtile * 16 + lo) * K + hi * 8;
    const _Float16* bp  = Bp + ((size_t)ntile * nkt) * 512 + (size_t)lane * 16;

    for (int kk = 0; kk < nkt; ++kk) {
        Frag16 a, b;
        a.q[0] = *(const uint4*)(arp + kk * 32);
        a.q[1] = *(const uint4*)(arp + kk * 32 + 16);
        b.q[0] = *(const uint4*)(bp + (size_t)kk * 512);
        b.q[1] = *(const uint4*)(bp + (size_t)kk * 512 + 8);
        acc.v = __builtin_amdgcn_wmma_f32_16x16x32_f16(
            false, a.v, false, b.v, (short)0, acc.v, false, false);
    }

    if (mode == 0) {
        float* cp = C + (size_t)(mtile * 16 + hi * 8) * N + col;
#pragma unroll
        for (int j = 0; j < 8; ++j) cp[(size_t)j * N] = acc.f[j];
    } else {
        // rows m = mtile*16 + r are consecutive t within one b (T % 16 == 0)
        int bidx = (mtile * 16) >> 11;       // m / TLEN
        int t0   = (mtile * 16) & (TLEN - 1);
        float* cp = C + ((size_t)(t0 + hi * 8) * BSZ + bidx) * N + col;
#pragma unroll
        for (int j = 0; j < 8; ++j) cp[(size_t)j * BSZ * N] = acc.f[j];
    }
}

// ---------------------------------------------------------------------------
// TDM: issue async 1-row tile load of n f32 elements, global -> LDS.
// Descriptor per CDNA5 ISA ch.8: group0 {count,lds_addr,global_addr,type=2},
// group1 {data_size=4B, tensor_dim0=n, tensor_dim1=1, tile_dim0=n, tile_dim1=1}.
// 6-arg builtin form (clang-23 / therock-10.0 headers).
// ---------------------------------------------------------------------------
#if HAVE_TDM
__device__ __forceinline__ void tdm_load_f32_1d(unsigned int lds_off,
                                                const float* src,
                                                unsigned int n) {
    unsigned long long ga = (unsigned long long)(uintptr_t)src;
    v4u g0;
    g0[0] = 1u;                                            // count=1 (valid load)
    g0[1] = lds_off;                                       // lds_addr (bytes)
    g0[2] = (unsigned int)(ga & 0xFFFFFFFFu);              // global_addr[31:0]
    g0[3] = (unsigned int)((ga >> 32) & 0x1FFFFFFu)        // global_addr[56:32]
          | (2u << 30);                                    // type = 2 (image)
    v8i_ g1;
    g1[0] = (int)(2u << 16);                               // data_size = 4 bytes
    g1[1] = (int)((n & 0xFFFFu) << 16);                    // tensor_dim0[15:0]
    g1[2] = (int)(((n >> 16) & 0xFFFFu) | (1u << 16));     // dim0[31:16] | tensor_dim1=1
    g1[3] = (int)((n & 0xFFFFu) << 16);                    // tile_dim0 = n
    g1[4] = 1;                                             // tile_dim1=1, tile_dim2=0
    g1[5] = (int)n;                                        // tensor_dim0_stride[31:0]
    g1[6] = 0;
    g1[7] = 0;
    v4i_ z4 = {0, 0, 0, 0};
    v8i_ z8 = {0, 0, 0, 0, 0, 0, 0, 0};
    __builtin_amdgcn_tensor_load_to_lds(g0, g1, z4, z4, z8, 0);
}
#endif

// ---------------------------------------------------------------------------
// LSTM recurrence. 4 blocks x 512 threads (16 waves). Block bb owns batch
// rows [bb*16, bb*16+16). Wave w owns h-columns [w*16, w*16+16).
// xg (layout [T, B, 4H] f32) is double-buffered in LDS via the Tensor Data
// Mover, prefetched one step ahead; h double-buffered in LDS (f16); c
// persistent in f32 registers (WMMA C-tile layout).
// ---------------------------------------------------------------------------
__global__ __launch_bounds__(512)
void lstm_seq_kernel(const float* __restrict__ xgT,
                     const _Float16* __restrict__ Whp,
                     _Float16* __restrict__ hs) {
    __shared__ _Float16 hbuf[2][16 * HID];
    __shared__ float    xstage[2][XSTAGE_ELEMS];

    const int lane = threadIdx.x & 31;
    const int w    = threadIdx.x >> 5;   // 0..15
    const int bb   = blockIdx.x;         // 0..3
    const int hi   = lane >> 4;
    const int lo   = lane & 15;
    const int colh = w * 16 + lo;        // owned h column (C-tile N)
    const int rbase = hi * 8;            // C-tile M base for element j

    for (int i = threadIdx.x; i < 16 * HID; i += blockDim.x)
        hbuf[0][i] = (_Float16)0.f;

#if HAVE_TDM
    const unsigned int xs_base = (unsigned int)(uintptr_t)&xstage[0][0];
    if (w == 0) {
        tdm_load_f32_1d(xs_base,
                        xgT + (size_t)bb * 16 * G4H,   // t = 0 slice
                        XSTAGE_ELEMS);
        __builtin_amdgcn_s_wait_tensorcnt(0);
    }
#else
    for (int i = threadIdx.x; i < XSTAGE_ELEMS; i += blockDim.x)
        xstage[0][i] = xgT[(size_t)bb * 16 * G4H + i];
#endif

    AccU c;
#pragma unroll
    for (int j = 0; j < 8; ++j) c.f[j] = 0.f;

    __syncthreads();

    for (int t = 0; t < TLEN; ++t) {
        // prefetch xg slice for t+1 into the other staging buffer (overlaps
        // with this step's WMMA work; completion waited before the barrier)
#if HAVE_TDM
        if (w == 0 && t + 1 < TLEN) {
            tdm_load_f32_1d(xs_base + ((unsigned)((t + 1) & 1)) * (XSTAGE_ELEMS * 4u),
                            xgT + ((size_t)(t + 1) * BSZ + bb * 16) * G4H,
                            XSTAGE_ELEMS);
        }
#endif
        const float*    xs   = &xstage[t & 1][0];
        const _Float16* hcur = hbuf[t & 1];

        // seed gate accumulators from LDS-staged xg
        AccU g[4];
#pragma unroll
        for (int gi = 0; gi < 4; ++gi) {
#pragma unroll
            for (int j = 0; j < 8; ++j)
                g[gi].f[j] = xs[(rbase + j) * G4H + gi * HID + colh];
        }

        // g += h @ Wh   (K = 256 -> 8 WMMA steps, 4 gate tiles each)
        const _Float16* arp = hcur + lo * HID + hi * 8;
#pragma unroll
        for (int kk = 0; kk < 8; ++kk) {
            Frag16 a;
            a.q[0] = *(const uint4*)(arp + kk * 32);
            a.q[1] = *(const uint4*)(arp + kk * 32 + 16);
#pragma unroll
            for (int gi = 0; gi < 4; ++gi) {
                const int ntile = gi * 16 + w;
                const _Float16* bp = Whp + ((size_t)ntile * 8 + kk) * 512
                                   + (size_t)lane * 16;
                Frag16 b;
                b.q[0] = *(const uint4*)bp;
                b.q[1] = *(const uint4*)(bp + 8);
                g[gi].v = __builtin_amdgcn_wmma_f32_16x16x32_f16(
                    false, a.v, false, b.v, (short)0, g[gi].v, false, false);
            }
        }

        // elementwise LSTM cell update (f32)
        _Float16 hnew[8];
#pragma unroll
        for (int j = 0; j < 8; ++j) {
            float fg = 1.f / (1.f + __expf(-g[0].f[j]));
            float ig = 1.f / (1.f + __expf(-g[1].f[j]));
            float og = 1.f / (1.f + __expf(-g[2].f[j]));
            float ct = tanhf(g[3].f[j]);
            c.f[j] = c.f[j] * fg + ct * ig;
            hnew[j] = (_Float16)(og * tanhf(c.f[j]));
        }

        // publish h to next LDS buffer + hs (f16) for the output GEMM
        _Float16* hnext = hbuf[(t + 1) & 1];
#pragma unroll
        for (int j = 0; j < 8; ++j) {
            int r = rbase + j;
            hnext[r * HID + colh] = hnew[j];
            hs[((size_t)(bb * 16 + r) * TLEN + t) * HID + colh] = hnew[j];
        }

#if HAVE_TDM
        if (w == 0) __builtin_amdgcn_s_wait_tensorcnt(0);
#else
        if (t + 1 < TLEN) {
            __syncthreads();
            for (int i = threadIdx.x; i < XSTAGE_ELEMS; i += blockDim.x)
                xstage[(t + 1) & 1][i] =
                    xgT[((size_t)(t + 1) * BSZ + bb * 16) * G4H + i];
        }
#endif
        __syncthreads();
    }
}

// ---------------------------------------------------------------------------
extern "C" void kernel_launch(void* const* d_in, const int* in_sizes, int n_in,
                              void* d_out, int out_size, void* d_ws, size_t ws_size,
                              hipStream_t stream) {
    const float* X    = (const float*)d_in[0];
    const float* Wx   = (const float*)d_in[1];
    const float* Wh   = (const float*)d_in[2];
    const float* b    = (const float*)d_in[3];
    const float* Wout = (const float*)d_in[4];
    const float* bout = (const float*)d_in[5];
    float* out = (float*)d_out;

    char* ws = (char*)d_ws;
    size_t off = 0;
    float* xgT = (float*)(ws + off);          off += (size_t)BSZ * TLEN * G4H * sizeof(float);
    _Float16* hsb = (_Float16*)(ws + off);    off += (size_t)BSZ * TLEN * HID * sizeof(_Float16);
    _Float16* Xh  = (_Float16*)(ws + off);    off += (size_t)BSZ * TLEN * DIN * sizeof(_Float16);
    _Float16* Wxp = (_Float16*)(ws + off);    off += (size_t)DIN * G4H * sizeof(_Float16);
    _Float16* Whp = (_Float16*)(ws + off);    off += (size_t)HID * G4H * sizeof(_Float16);
    _Float16* Wop = (_Float16*)(ws + off);    off += (size_t)HID * OUTD * sizeof(_Float16);

    // 1) convert X to f16
    int nX = BSZ * TLEN * DIN;
    cvt_f16_kernel<<<(nX + 255) / 256, 256, 0, stream>>>(X, Xh, nX);

    // 2) pack weights into WMMA B-fragment order (f16)
    int pWx = (DIN / 32) * (G4H / 16) * 32;
    int pWh = (HID / 32) * (G4H / 16) * 32;
    int pWo = (HID / 32) * (OUTD / 16) * 32;
    pack_b_kernel<<<(pWx + 255) / 256, 256, 0, stream>>>(Wx,   Wxp, DIN, G4H);
    pack_b_kernel<<<(pWh + 255) / 256, 256, 0, stream>>>(Wh,   Whp, HID, G4H);
    pack_b_kernel<<<(pWo + 255) / 256, 256, 0, stream>>>(Wout, Wop, HID, OUTD);

    // 3) xg = X @ Wx + b, written as [T, B, 4H] for contiguous TDM tiles
    gemm_wmma_kernel<<<dim3(BSZ * TLEN / 16, G4H / 16 / 8), 256, 0, stream>>>(
        Xh, Wxp, b, xgT, DIN, G4H, 1);

    // 4) sequential LSTM over T with TDM-prefetched xg; writes hs (f16)
    lstm_seq_kernel<<<dim3(BSZ / 16), 512, 0, stream>>>(xgT, Whp, hsb);

    // 5) out = hs @ Wout + bout  (M=131072, K=256, N=128)
    gemm_wmma_kernel<<<dim3(BSZ * TLEN / 16, OUTD / 16 / 8), 256, 0, stream>>>(
        hsb, Wop, bout, out, HID, OUTD, 0);
}
